// VectorizedResNet_62431644615067
// MI455X (gfx1250) — matrix operand order repaired and checked by reading, not compiled, not run on hardware
//
#include <hip/hip_runtime.h>

#define EPSV 1e-5f

typedef __attribute__((ext_vector_type(16))) __bf16        v16bf;
typedef __attribute__((ext_vector_type(8)))  float         v8f;
typedef __attribute__((ext_vector_type(8)))  unsigned int  v8u;

__device__ __forceinline__ unsigned short f2bf(float f) {
  unsigned int u = __builtin_bit_cast(unsigned int, f);
  u += 0x7FFFu + ((u >> 16) & 1u);          // round-to-nearest-even
  return (unsigned short)(u >> 16);
}
__device__ __forceinline__ float bf2f(unsigned short h) {
  unsigned int u = ((unsigned int)h) << 16;
  return __builtin_bit_cast(float, u);
}

// ---------------------------------------------------------------------------
// NCHW fp32 -> NHWC bf16 (stem input, shared across models)
// ---------------------------------------------------------------------------
__global__ void k_x_to_nhwc(const float* __restrict__ x, unsigned short* __restrict__ out,
                            int Bv, int C, int H, int W) {
  long long i = (long long)blockIdx.x * blockDim.x + threadIdx.x;
  long long total = (long long)Bv * C * H * W;
  if (i >= total) return;
  int c = (int)(i % C); long long t = i / C;
  int xx = (int)(t % W); t /= W;
  int yy = (int)(t % H); t /= H;
  int b = (int)t;
  out[i] = f2bf(x[(((long long)b * C + c) * H + yy) * W + xx]);
}

// ---------------------------------------------------------------------------
// BN-fold + pre-swizzle weights into WMMA A-fragment order.
// K ordering: k = (r*kw + s)*Cin + cin   (tap-major, channel-minor)
// Output layout: [m][cout_tile][k_chunk][lane(32)][e(16)] bf16, where
// lane = half*16 + row, and element e maps to
//   k = kc*32 + 16*half + e     (contiguous k order, matches B)
// so the conv kernel loads one contiguous 32B fragment per lane per chunk.
// Also emits bias[m][cout] = b - mean*scale.
// ---------------------------------------------------------------------------
__global__ void k_prep_wswz(const float* __restrict__ w, const float* __restrict__ g,
                            const float* __restrict__ bb, const float* __restrict__ mm,
                            const float* __restrict__ vv,
                            unsigned short* __restrict__ wout, float* __restrict__ bout,
                            int Mv, int Cout, int Cin, int khw, int K, int nchunks) {
  long long i = (long long)blockIdx.x * blockDim.x + threadIdx.x;
  int CT = Cout >> 4;
  long long total = (long long)Mv * CT * nchunks * 512;
  if (i >= total) return;
  int e    = (int)(i & 15);
  int lane = (int)((i >> 4) & 31);
  long long t = i >> 9;
  int kc  = (int)(t % nchunks); t /= nchunks;
  int ctt = (int)(t % CT);
  int m   = (int)(t / CT);
  int row = lane & 15, half = lane >> 4;
  int cout = ctt * 16 + row;
  int mc = m * Cout + cout;
  float scale = g[mc] * rsqrtf(vv[mc] + EPSV);
  int k = kc * 32 + half * 16 + e;
  float val = 0.f;
  if (k < K) {
    int rs = k / Cin, cin = k - rs * Cin;
    val = w[((long long)mc * Cin + cin) * khw + rs] * scale;
  }
  wout[i] = f2bf(val);
  if (kc == 0 && e == 0 && half == 0) bout[mc] = bb[mc] - mm[mc] * scale;
}

// ---------------------------------------------------------------------------
// Implicit-GEMM conv + bias (+residual) (+ReLU), bf16 in/out, fp32 WMMA accum.
// Block = 128 threads = 4 waves; one 16-pixel tile per block. Each wave owns
// `ctper` consecutive Cout tiles (2 for Cout>=128 -> 2 WMMAs per chunk over a
// shared B fragment). B (32K x 16N im2col patch) is staged in a DOUBLE-
// buffered LDS tile: tile (kc+1) is filled while tile (kc) feeds the WMMAs,
// one barrier per chunk. Fast path (Cin%32==0): each K-chunk is one (r,s)
// tap + 32 contiguous channels -> u64 vector global loads + u64 LDS stores.
// ---------------------------------------------------------------------------
__global__ void __launch_bounds__(128)
k_conv_wmma(const unsigned short* __restrict__ in, unsigned long long in_mstride,
            const unsigned short* __restrict__ wswz, const float* __restrict__ bias,
            const unsigned short* __restrict__ res, unsigned short* __restrict__ out,
            int Bv, int Cin, int Hin, int Win, int Cout, int Hout, int Wout,
            int kh, int kw, int stride, int pad, int K, int nchunks, int ctper, int relu) {
  __shared__ __attribute__((aligned(64))) unsigned short lds_b[2][16 * 32];

  const int tid   = threadIdx.x;
  const int lane  = tid & 31;
  const int wv    = tid >> 5;
  const int ptile = blockIdx.x;
  const int mb    = blockIdx.z;
  const int m     = mb / Bv;
  const int b     = mb % Bv;
  const int HW    = Hout * Wout;
  const int CT    = Cout >> 4;
  const int ct0   = blockIdx.y * 4 * ctper + wv * ctper;
  const int half  = lane >> 4;
  const int l16   = lane & 15;
  const bool fast = (Cin & 31) == 0;
  const bool two  = (ctper == 2);

  const unsigned short* inb = in + (unsigned long long)m * in_mstride
                                 + (unsigned long long)b * Hin * Win * Cin;
  const unsigned short* wb0 = wswz
      + ((unsigned long long)m * CT + ct0) * ((unsigned long long)nchunks * 512)
      + (unsigned long long)lane * 16;
  const unsigned short* wb1 = wb0 + (unsigned long long)nchunks * 512;

  // fast-path fill coordinates for this thread: n = tid>>3, j = tid&7
  const int fn = tid >> 3, fj = tid & 7;
  const int fp = ptile * 16 + fn;
  const bool fvalid = fp < HW;
  const int foy = fvalid ? (fp / Wout) : 0;
  const int fox = fvalid ? (fp - foy * Wout) : 0;

  // stage im2col tile for chunk kc into LDS buffer `buf`
  auto fill = [&](int kc, int buf) {
    if (fast) {
      int kbase = kc << 5;
      int rs = kbase / Cin, cinb = kbase - rs * Cin;
      int r = rs / kw, s = rs - r * kw;
      int iy = foy * stride - pad + r;
      int ix = fox * stride - pad + s;
      unsigned long long v = 0ull;
      if (fvalid && iy >= 0 && iy < Hin && ix >= 0 && ix < Win)
        v = *(const unsigned long long*)(inb + ((long long)iy * Win + ix) * Cin + cinb + (fj << 2));
      *(unsigned long long*)(&lds_b[buf][(fn << 5) + (fj << 2)]) = v;
    } else {
      for (int i = tid; i < 512; i += 128) {
        int n = i >> 5, kk = i & 31;
        int k = (kc << 5) + kk;
        float v = 0.f;
        int p = ptile * 16 + n;
        if (k < K && p < HW) {
          int rs = k / Cin, cin = k - rs * Cin;
          int r = rs / kw, s = rs - r * kw;
          int oy = p / Wout, ox = p - oy * Wout;
          int iy = oy * stride - pad + r;
          int ix = ox * stride - pad + s;
          if (iy >= 0 && iy < Hin && ix >= 0 && ix < Win)
            v = bf2f(inb[((long long)iy * Win + ix) * Cin + cin]);
        }
        lds_b[buf][n * 32 + kk] = f2bf(v);
      }
    }
  };

  v8f acc0 = {}, acc1 = {};

  fill(0, 0);
  __syncthreads();

  for (int kc = 0; kc < nchunks; ++kc) {
    const int cur = kc & 1;
    // prefetch next tile into the other LDS buffer (no barrier needed: the
    // end-of-iteration barrier of chunk kc-1 already retired all reads of it)
    if (kc + 1 < nchunks) {
      __builtin_prefetch(wb0 + (kc + 1) * 512, 0, 0);
      fill(kc + 1, cur ^ 1);
    }

    // ---- A fragments: one contiguous 32B load per lane (pre-swizzled) ----
    v8u au0 = *(const v8u*)(wb0 + (unsigned long long)kc * 512);
    // ---- B fragment: 16 contiguous bf16 from LDS ----
    v8u bu = *(const v8u*)(&lds_b[cur][l16 * 32 + (half << 4)]);
    v16bf bmat = __builtin_bit_cast(v16bf, bu);

    acc0 = __builtin_amdgcn_wmma_f32_16x16x32_bf16(
        false, __builtin_bit_cast(v16bf, au0), false, bmat, (short)0, acc0, false, false);
    if (two) {
      v8u au1 = *(const v8u*)(wb1 + (unsigned long long)kc * 512);
      acc1 = __builtin_amdgcn_wmma_f32_16x16x32_bf16(
          false, __builtin_bit_cast(v16bf, au1), false, bmat, (short)0, acc1, false, false);
    }
    __syncthreads();
  }

  // ---- epilogue: C/D layout => VGPR r holds row (half*8 + r), col = lane&15
  int p = ptile * 16 + l16;
  if (p < HW) {
    unsigned long long obase = ((unsigned long long)mb * HW + p) * (unsigned long long)Cout;
    const float* bs = bias + (unsigned long long)m * Cout;
    for (int tt = 0; tt < ctper; ++tt) {
      v8f acc = tt ? acc1 : acc0;
      int cb = (ct0 + tt) * 16 + half * 8;
      #pragma unroll
      for (int r = 0; r < 8; ++r) {
        int co = cb + r;
        float v = acc[r] + bs[co];
        if (res) v += bf2f(res[obase + co]);
        if (relu) v = fmaxf(v, 0.f);
        out[obase + co] = f2bf(v);
      }
    }
  }
}

// ---------------------------------------------------------------------------
// 3x3 stride-2 pad-1 maxpool, bf16 NHWC
// ---------------------------------------------------------------------------
__global__ void k_maxpool(const unsigned short* __restrict__ in, unsigned short* __restrict__ out,
                          int MBv, int C, int Hin, int Win, int Hout, int Wout) {
  long long i = (long long)blockIdx.x * blockDim.x + threadIdx.x;
  long long total = (long long)MBv * Hout * Wout * C;
  if (i >= total) return;
  int c = (int)(i % C); long long t = i / C;
  int ox = (int)(t % Wout); t /= Wout;
  int oy = (int)(t % Hout); t /= Hout;
  int mb = (int)t;
  const unsigned short* ib = in + (long long)mb * Hin * Win * C;
  float best = -3.4e38f;
  for (int r = 0; r < 3; ++r) {
    int iy = oy * 2 - 1 + r;
    if (iy < 0 || iy >= Hin) continue;
    for (int s = 0; s < 3; ++s) {
      int ix = ox * 2 - 1 + s;
      if (ix < 0 || ix >= Win) continue;
      best = fmaxf(best, bf2f(ib[((long long)iy * Win + ix) * C + c]));
    }
  }
  out[i] = f2bf(best);
}

// ---------------------------------------------------------------------------
// Global avg pool (7x7) + FC 512->2, fp32 out [M,B,2]
// ---------------------------------------------------------------------------
__global__ void k_avgfc(const unsigned short* __restrict__ h, const float* __restrict__ fw,
                        const float* __restrict__ fb, float* __restrict__ out, int Bv, int C) {
  int mb = blockIdx.x;
  int m = mb / Bv;
  int tid = threadIdx.x;
  __shared__ float s0[256], s1[256];
  const unsigned short* hb = h + (unsigned long long)mb * 49 * C;
  float a0 = 0.f, a1 = 0.f;
  for (int c = tid; c < C; c += 256) {
    float s = 0.f;
    for (int p = 0; p < 49; ++p) s += bf2f(hb[p * C + c]);
    s *= (1.f / 49.f);
    a0 += s * fw[((long long)m * 2 + 0) * C + c];
    a1 += s * fw[((long long)m * 2 + 1) * C + c];
  }
  s0[tid] = a0; s1[tid] = a1;
  __syncthreads();
  for (int st = 128; st > 0; st >>= 1) {
    if (tid < st) { s0[tid] += s0[tid + st]; s1[tid] += s1[tid + st]; }
    __syncthreads();
  }
  if (tid == 0) {
    out[mb * 2 + 0] = s0[0] + fb[m * 2 + 0];
    out[mb * 2 + 1] = s1[0] + fb[m * 2 + 1];
  }
}

// ---------------------------------------------------------------------------
// Host orchestration
// ---------------------------------------------------------------------------
extern "C" void kernel_launch(void* const* d_in, const int* in_sizes, int n_in,
                              void* d_out, int out_size, void* d_ws, size_t ws_size,
                              hipStream_t stream) {
  (void)in_sizes; (void)n_in; (void)out_size; (void)ws_size;
  const int M = 8, B = 16;

  int idx = 0;
  const float* x = (const float*)d_in[idx++];

  struct CD {
    const float *w, *g, *b, *mm, *v;
    int Cout, Cin, kh, kw, K, nchunks, ctper;
    unsigned short* wbf; float* bias;
  };
  CD cd[20]; int nc = 0;
  auto grab = [&](int Cout, int Cin, int kh, int kw) -> int {
    CD d{};
    d.w  = (const float*)d_in[idx++];
    d.g  = (const float*)d_in[idx++];
    d.b  = (const float*)d_in[idx++];
    d.mm = (const float*)d_in[idx++];
    d.v  = (const float*)d_in[idx++];
    d.Cout = Cout; d.Cin = Cin; d.kh = kh; d.kw = kw;
    d.K = Cin * kh * kw;
    d.nchunks = (d.K + 31) / 32;
    d.ctper = (Cout >= 128) ? 2 : 1;
    cd[nc] = d; return nc++;
  };

  int c_stem = grab(64, 3, 7, 7);
  const int chan[4] = {64, 128, 256, 512};
  int cidA[4][2], cidB[4][2], cidD[4] = {-1, -1, -1, -1};
  int inc = 64;
  for (int si = 0; si < 4; ++si) {
    for (int bi = 0; bi < 2; ++bi) {
      int c = chan[si];
      int ci = (bi == 0) ? inc : c;
      cidA[si][bi] = grab(c, ci, 3, 3);
      cidB[si][bi] = grab(c, c, 3, 3);
      if (bi == 0 && si > 0) cidD[si] = grab(c, inc, 1, 1);
    }
    inc = chan[si];
  }
  const float* fcw = (const float*)d_in[idx++];
  const float* fcb = (const float*)d_in[idx++];

  // ---- bump-allocate workspace ----
  char* wp_ = (char*)d_ws;
  auto alloc = [&](size_t bytes) -> void* {
    void* r = (void*)wp_;
    wp_ += (bytes + 255) & ~(size_t)255;
    return r;
  };
  unsigned short* xnhwc = (unsigned short*)alloc((size_t)B * 224 * 224 * 3 * 2);
  for (int i = 0; i < nc; ++i) {
    size_t welems = (size_t)M * (cd[i].Cout / 16) * cd[i].nchunks * 512;
    cd[i].wbf  = (unsigned short*)alloc(welems * 2);
    cd[i].bias = (float*)alloc((size_t)M * cd[i].Cout * 4);
  }
  unsigned short* A1 = (unsigned short*)alloc((size_t)M * B * 112 * 112 * 64 * 2);
  size_t wmax = (size_t)M * B * 56 * 56 * 64 * 2;
  unsigned short* W0 = (unsigned short*)alloc(wmax);
  unsigned short* W1 = (unsigned short*)alloc(wmax);
  unsigned short* W2 = (unsigned short*)alloc(wmax);
  unsigned short* W3 = (unsigned short*)alloc(wmax);

  // ---- prep: input layout + BN-folded swizzled bf16 weights ----
  {
    long long tot = (long long)B * 3 * 224 * 224;
    k_x_to_nhwc<<<dim3((unsigned)((tot + 255) / 256)), 256, 0, stream>>>(x, xnhwc, B, 3, 224, 224);
  }
  for (int i = 0; i < nc; ++i) {
    long long tot = (long long)M * (cd[i].Cout / 16) * cd[i].nchunks * 512;
    k_prep_wswz<<<dim3((unsigned)((tot + 255) / 256)), 256, 0, stream>>>(
        cd[i].w, cd[i].g, cd[i].b, cd[i].mm, cd[i].v, cd[i].wbf, cd[i].bias,
        M, cd[i].Cout, cd[i].Cin, cd[i].kh * cd[i].kw, cd[i].K, cd[i].nchunks);
  }

  auto conv = [&](const unsigned short* in, size_t in_mstride, int Hin, int Win, int ci_,
                  int Hout, int Wout, const CD& d, int stride, int pad,
                  const unsigned short* res, int relu, unsigned short* out) {
    int HW = Hout * Wout;
    dim3 grid((unsigned)((HW + 15) / 16), (unsigned)(d.Cout / (64 * d.ctper)), (unsigned)(M * B));
    k_conv_wmma<<<grid, 128, 0, stream>>>(in, (unsigned long long)in_mstride,
        d.wbf, d.bias, res, out, B, ci_, Hin, Win, d.Cout, Hout, Wout,
        d.kh, d.kw, stride, pad, d.K, d.nchunks, d.ctper, relu);
  };

  // stem: 7x7/2 conv + bn + relu, then 3x3/2 maxpool
  conv(xnhwc, 0 /*shared across models*/, 224, 224, 3, 112, 112, cd[c_stem], 2, 3, nullptr, 1, A1);
  {
    long long tot = (long long)M * B * 56 * 56 * 64;
    k_maxpool<<<dim3((unsigned)((tot + 255) / 256)), 256, 0, stream>>>(A1, W0, M * B, 64, 112, 112, 56, 56);
  }

  int Hc = 56, Cc = 64;
  unsigned short* cur = W0;
  for (int si = 0; si < 4; ++si) {
    int c = chan[si];
    int s = (si == 0) ? 1 : 2;
    int Ho = Hc / s;
    size_t ims_in = (size_t)B * Hc * Hc * Cc;
    size_t ims_o  = (size_t)B * Ho * Ho * c;

    // block 0 (strided; downsample path for si>0)
    unsigned short* resb;
    if (si > 0) {
      conv(cur, ims_in, Hc, Hc, Cc, Ho, Ho, cd[cidD[si]], s, 0, nullptr, 0, W2);
      resb = W2;
    } else {
      resb = cur;                         // identity residual (W0)
    }
    conv(cur, ims_in, Hc, Hc, Cc, Ho, Ho, cd[cidA[si][0]], s, 1, nullptr, 1, W1);
    unsigned short* out0 = (si > 0) ? W3 : W2;
    conv(W1, ims_o, Ho, Ho, c, Ho, Ho, cd[cidB[si][0]], 1, 1, resb, 1, out0);

    // block 1 (stride 1, identity residual = out0)
    conv(out0, ims_o, Ho, Ho, c, Ho, Ho, cd[cidA[si][1]], 1, 1, nullptr, 1, W1);
    conv(W1, ims_o, Ho, Ho, c, Ho, Ho, cd[cidB[si][1]], 1, 1, out0, 1, W0);

    cur = W0; Hc = Ho; Cc = c;
  }

  // head: global avg pool + fc -> fp32 [M,B,2]
  k_avgfc<<<dim3((unsigned)(M * B)), 256, 0, stream>>>(cur, fcw, fcb, (float*)d_out, B, 512);
}
